// Seq2Seq_90400471646291
// MI455X (gfx1250) — matrix operand order repaired and checked by reading, not compiled, not run on hardware
//
#include <hip/hip_runtime.h>
#include <stdint.h>
#include <stddef.h>

// ---------------------------------------------------------------------------
// Problem constants (from the reference)
// ---------------------------------------------------------------------------
#define S_LEN   256
#define T_LEN   128
#define BATCH   256
#define FDIM    128
#define HE_DIM  512
#define HD_DIM  768
#define DESC_DIM 384
#define CH_DIM  (HD_DIM - HE_DIM)   // 256

typedef __bf16 bf16_t;
typedef __attribute__((ext_vector_type(16))) __bf16 v16bf;
typedef __attribute__((ext_vector_type(8)))  __bf16 v8bf;
typedef __attribute__((ext_vector_type(8)))  float  v8f;

// ---------------------------------------------------------------------------
// bf16 helpers (bit-level, no __bf16 arithmetic so host pass stays happy)
// ---------------------------------------------------------------------------
__device__ __forceinline__ unsigned short f32_to_bf16_bits(float x) {
  union { float f; unsigned u; } v; v.f = x;
  unsigned r = v.u + 0x7FFFu + ((v.u >> 16) & 1u);   // round-to-nearest-even
  return (unsigned short)(r >> 16);
}

// ---------------------------------------------------------------------------
// Elementwise utility kernels
// ---------------------------------------------------------------------------
__global__ void k_f32_to_bf16(const float* __restrict__ in,
                              unsigned short* __restrict__ out, int n) {
  int i = blockIdx.x * blockDim.x + threadIdx.x;
  if (i < n) out[i] = f32_to_bf16_bits(in[i]);
}

__global__ void k_zero_u32(unsigned int* __restrict__ p, int nWords) {
  int i = blockIdx.x * blockDim.x + threadIdx.x;
  if (i < nWords) p[i] = 0u;
}

// ---------------------------------------------------------------------------
// Fused WMMA GEMM:
//   Z[256 x N] = Xbf[256 x Kx] * Wx^T  (+ Hbf[256 x Kh] * Wh^T)  (+ bias0 + bias1)
// Wx is [N x Kx] row-major bf16 (weight layout of the reference), likewise Wh.
// grid = ((N+255)/256, 256/32), block = 128 (4 wave32s).
// Each wave computes a 32x64 strip: 2 A-tiles x 4 B-tiles, 8 WMMAs per K-step.
// ---------------------------------------------------------------------------
__device__ __forceinline__ v16bf load_a_k32(const bf16_t* __restrict__ rowPtr,
                                            int k0, int laneHalf) {
  // A 16x32 bf16 layout: lane<16 -> K in {0..7}U{16..23}, lane>=16 -> +8
  const bf16_t* p = rowPtr + k0 + laneHalf * 8;
  v8bf lo = *(const v8bf*)(p);
  v8bf hi = *(const v8bf*)(p + 16);
  v16bf r;
#pragma unroll
  for (int i = 0; i < 8; ++i) { r[i] = lo[i]; r[i + 8] = hi[i]; }
  return r;
}

__device__ __forceinline__ v16bf load_b_k32(const bf16_t* __restrict__ rowPtr,
                                            int k0, int laneHalf) {
  // B 32x16 bf16 layout: lane<16 -> K in {0..7}U{16..23}, lane>=16 -> +8
  const bf16_t* p = rowPtr + k0 + laneHalf * 8;
  v8bf lo = *(const v8bf*)(p);
  v8bf hi = *(const v8bf*)(p + 16);
  v16bf r;
#pragma unroll
  for (int i = 0; i < 8; ++i) { r[i] = lo[i]; r[i + 8] = hi[i]; }
  return r;
}

__global__ __launch_bounds__(128, 1)
void k_gemm_wmma(const bf16_t* __restrict__ X, int Kx, const bf16_t* __restrict__ Wx,
                 const bf16_t* __restrict__ H, int Kh, const bf16_t* __restrict__ Wh,
                 const float* __restrict__ bias0, const float* __restrict__ bias1,
                 float* __restrict__ Z, int N)
{
  const int lane     = threadIdx.x & 31;
  const int wave     = threadIdx.x >> 5;
  const int lp       = lane & 15;
  const int laneHalf = lane >> 4;
  const int rowBase  = blockIdx.y << 5;                 // 32 rows per WG
  const int col0     = (blockIdx.x << 8) + (wave << 6); // 64 cols per wave
  if (col0 >= N) return;                                // wave-uniform exit

  v8f acc[2][4];
#pragma unroll
  for (int ct = 0; ct < 4; ++ct) {
    const int n = col0 + ct * 16 + lp;
    float bval = 0.0f;
    if (bias0) bval += bias0[n];
    if (bias1) bval += bias1[n];
#pragma unroll
    for (int rt = 0; rt < 2; ++rt)
#pragma unroll
      for (int e = 0; e < 8; ++e) acc[rt][ct][e] = bval;
  }

  // ---- X @ Wx^T ----
  {
    const bf16_t* xr0 = X + (size_t)(rowBase + lp) * Kx;
    const bf16_t* xr1 = X + (size_t)(rowBase + 16 + lp) * Kx;
    for (int k0 = 0; k0 < Kx; k0 += 32) {
      v16bf a0 = load_a_k32(xr0, k0, laneHalf);
      v16bf a1 = load_a_k32(xr1, k0, laneHalf);
#pragma unroll
      for (int ct = 0; ct < 4; ++ct) {
        v16bf b = load_b_k32(Wx + (size_t)(col0 + ct * 16 + lp) * Kx, k0, laneHalf);
        acc[0][ct] = __builtin_amdgcn_wmma_f32_16x16x32_bf16(
            false, a0, false, b, (short)0, acc[0][ct], false, false);
        acc[1][ct] = __builtin_amdgcn_wmma_f32_16x16x32_bf16(
            false, a1, false, b, (short)0, acc[1][ct], false, false);
      }
    }
  }

  // ---- + H @ Wh^T (optional) ----
  if (Wh) {
    const bf16_t* hr0 = H + (size_t)(rowBase + lp) * Kh;
    const bf16_t* hr1 = H + (size_t)(rowBase + 16 + lp) * Kh;
    for (int k0 = 0; k0 < Kh; k0 += 32) {
      v16bf a0 = load_a_k32(hr0, k0, laneHalf);
      v16bf a1 = load_a_k32(hr1, k0, laneHalf);
#pragma unroll
      for (int ct = 0; ct < 4; ++ct) {
        v16bf b = load_b_k32(Wh + (size_t)(col0 + ct * 16 + lp) * Kh, k0, laneHalf);
        acc[0][ct] = __builtin_amdgcn_wmma_f32_16x16x32_bf16(
            false, a0, false, b, (short)0, acc[0][ct], false, false);
        acc[1][ct] = __builtin_amdgcn_wmma_f32_16x16x32_bf16(
            false, a1, false, b, (short)0, acc[1][ct], false, false);
      }
    }
  }

  // ---- store C: reg e of lane -> row = base + e + 8*laneHalf, col = lp ----
#pragma unroll
  for (int rt = 0; rt < 2; ++rt) {
    const int mBase = rowBase + rt * 16 + laneHalf * 8;
#pragma unroll
    for (int ct = 0; ct < 4; ++ct) {
      const int n = col0 + ct * 16 + lp;
#pragma unroll
      for (int e = 0; e < 8; ++e)
        Z[(size_t)(mBase + e) * N + n] = acc[rt][ct][e];
    }
  }
}

// ---------------------------------------------------------------------------
// LSTM cell pointwise: gates (i,f,g,o) from z[B x 4H]; fp32 c, bf16 h out.
// ---------------------------------------------------------------------------
__global__ void k_lstm_cell(const float* __restrict__ z, float* __restrict__ c,
                            unsigned short* __restrict__ h_bf, int Hdim)
{
  int idx = blockIdx.x * blockDim.x + threadIdx.x;
  if (idx >= BATCH * Hdim) return;
  int b = idx / Hdim;
  int j = idx - b * Hdim;
  const float* zr = z + (size_t)b * 4 * Hdim;
  float zi = zr[j];
  float zf = zr[Hdim + j];
  float zg = zr[2 * Hdim + j];
  float zo = zr[3 * Hdim + j];
  float ig = 1.0f / (1.0f + __expf(-zi));
  float fg = 1.0f / (1.0f + __expf(-zf));
  float gg = tanhf(zg);
  float og = 1.0f / (1.0f + __expf(-zo));
  float cn = fg * c[idx] + ig * gg;
  c[idx] = cn;
  h_bf[idx] = f32_to_bf16_bits(og * tanhf(cn));
}

// ---------------------------------------------------------------------------
// Build decoder initial state for one layer:
//   dec_h = concat(bf16(ch), enc_h) ; dec_c = concat(0, enc_c)
// ---------------------------------------------------------------------------
__global__ void k_build_dec_state(const float* __restrict__ ch,
                                  const unsigned short* __restrict__ enc_h,
                                  const float* __restrict__ enc_c,
                                  unsigned short* __restrict__ dec_h,
                                  float* __restrict__ dec_c)
{
  int idx = blockIdx.x * blockDim.x + threadIdx.x;
  if (idx >= BATCH * HD_DIM) return;
  int b = idx / HD_DIM;
  int j = idx - b * HD_DIM;
  if (j < CH_DIM) {
    dec_h[idx] = f32_to_bf16_bits(ch[b * CH_DIM + j]);
    dec_c[idx] = 0.0f;
  } else {
    dec_h[idx] = enc_h[b * HE_DIM + (j - CH_DIM)];
    dec_c[idx] = enc_c[b * HE_DIM + (j - CH_DIM)];
  }
}

// ---------------------------------------------------------------------------
// Host orchestration
// ---------------------------------------------------------------------------
extern "C" void kernel_launch(void* const* d_in, const int* in_sizes, int n_in,
                              void* d_out, int out_size, void* d_ws, size_t ws_size,
                              hipStream_t stream)
{
  (void)in_sizes; (void)n_in; (void)out_size; (void)ws_size;

  const float* src   = (const float*)d_in[0];
  /* d_in[1] = trg: unused (teacher_forcing_ratio == 0; only its length T matters) */
  const float* descp = (const float*)d_in[2];
  const float* eWih0 = (const float*)d_in[3];
  const float* eWhh0 = (const float*)d_in[4];
  const float* ebih0 = (const float*)d_in[5];
  const float* ebhh0 = (const float*)d_in[6];
  const float* eWih1 = (const float*)d_in[7];
  const float* eWhh1 = (const float*)d_in[8];
  const float* ebih1 = (const float*)d_in[9];
  const float* ebhh1 = (const float*)d_in[10];
  const float* dWih0 = (const float*)d_in[11];
  const float* dWhh0 = (const float*)d_in[12];
  const float* dbih0 = (const float*)d_in[13];
  const float* dbhh0 = (const float*)d_in[14];
  const float* dWih1 = (const float*)d_in[15];
  const float* dWhh1 = (const float*)d_in[16];
  const float* dbih1 = (const float*)d_in[17];
  const float* dbhh1 = (const float*)d_in[18];
  const float* fcW   = (const float*)d_in[19];
  const float* fcb   = (const float*)d_in[20];
  const float* linW  = (const float*)d_in[21];
  const float* linb  = (const float*)d_in[22];
  float* out = (float*)d_out;

  // ---- workspace bump allocator (256B aligned) ----
  uintptr_t cur = (uintptr_t)d_ws;
  auto alloc = [&](size_t bytes) -> void* {
    cur = (cur + 255) & ~(uintptr_t)255;
    void* r = (void*)cur;
    cur += bytes;
    return r;
  };

  const int N4E = 4 * HE_DIM;  // 2048
  const int N4D = 4 * HD_DIM;  // 3072

  // bf16 buffers
  unsigned short* src_bf  = (unsigned short*)alloc((size_t)S_LEN * BATCH * FDIM * 2);
  unsigned short* wih0e   = (unsigned short*)alloc((size_t)N4E * FDIM   * 2);
  unsigned short* whh0e   = (unsigned short*)alloc((size_t)N4E * HE_DIM * 2);
  unsigned short* wih1e   = (unsigned short*)alloc((size_t)N4E * HE_DIM * 2);
  unsigned short* whh1e   = (unsigned short*)alloc((size_t)N4E * HE_DIM * 2);
  unsigned short* wih0d   = (unsigned short*)alloc((size_t)N4D * FDIM   * 2);
  unsigned short* whh0d   = (unsigned short*)alloc((size_t)N4D * HD_DIM * 2);
  unsigned short* wih1d   = (unsigned short*)alloc((size_t)N4D * HD_DIM * 2);
  unsigned short* whh1d   = (unsigned short*)alloc((size_t)N4D * HD_DIM * 2);
  unsigned short* fcw_bf  = (unsigned short*)alloc((size_t)FDIM * HD_DIM * 2);
  unsigned short* linw_bf = (unsigned short*)alloc((size_t)CH_DIM * DESC_DIM * 2);
  unsigned short* desc_bf = (unsigned short*)alloc((size_t)BATCH * DESC_DIM * 2);
  unsigned short* eh_bf   = (unsigned short*)alloc((size_t)2 * BATCH * HE_DIM * 2);
  unsigned short* dh_bf   = (unsigned short*)alloc((size_t)2 * BATCH * HD_DIM * 2);
  unsigned short* xfb_bf  = (unsigned short*)alloc((size_t)BATCH * FDIM * 2);
  // fp32 buffers
  float* ec    = (float*)alloc((size_t)2 * BATCH * HE_DIM * 4);
  float* dc    = (float*)alloc((size_t)2 * BATCH * HD_DIM * 4);
  float* zbuf  = (float*)alloc((size_t)BATCH * N4D * 4);
  float* chbuf = (float*)alloc((size_t)BATCH * CH_DIM * 4);

  auto conv = [&](const float* in, unsigned short* o, int n) {
    k_f32_to_bf16<<<dim3((unsigned)((n + 255) / 256)), dim3(256), 0, stream>>>(in, o, n);
  };
  auto gemm = [&](const unsigned short* X, int Kx, const unsigned short* Wx,
                  const unsigned short* H, int Kh, const unsigned short* Wh,
                  const float* b0, const float* b1, float* Zo, int N) {
    dim3 g((unsigned)((N + 255) / 256), (unsigned)(BATCH / 32));
    k_gemm_wmma<<<g, dim3(128), 0, stream>>>(
        (const bf16_t*)X, Kx, (const bf16_t*)Wx,
        (const bf16_t*)H, Kh, (const bf16_t*)Wh, b0, b1, Zo, N);
  };
  auto cell = [&](float* z, float* c, unsigned short* h, int Hdim) {
    int n = BATCH * Hdim;
    k_lstm_cell<<<dim3((unsigned)((n + 255) / 256)), dim3(256), 0, stream>>>(z, c, h, Hdim);
  };

  // ---- one-time (per launch) conversions: weights + src + desc -> bf16 ----
  conv(src,   src_bf,  S_LEN * BATCH * FDIM);
  conv(eWih0, wih0e,   N4E * FDIM);
  conv(eWhh0, whh0e,   N4E * HE_DIM);
  conv(eWih1, wih1e,   N4E * HE_DIM);
  conv(eWhh1, whh1e,   N4E * HE_DIM);
  conv(dWih0, wih0d,   N4D * FDIM);
  conv(dWhh0, whh0d,   N4D * HD_DIM);
  conv(dWih1, wih1d,   N4D * HD_DIM);
  conv(dWhh1, whh1d,   N4D * HD_DIM);
  conv(fcW,   fcw_bf,  FDIM * HD_DIM);
  conv(linW,  linw_bf, CH_DIM * DESC_DIM);
  conv(descp, desc_bf, BATCH * DESC_DIM);

  // ---- zero-init encoder h (bf16) and c (fp32) ----
  {
    int hw = 2 * BATCH * HE_DIM / 2;  // ushort pairs -> u32 words
    k_zero_u32<<<dim3((unsigned)((hw + 255) / 256)), dim3(256), 0, stream>>>(
        (unsigned int*)eh_bf, hw);
    int cw = 2 * BATCH * HE_DIM;
    k_zero_u32<<<dim3((unsigned)((cw + 255) / 256)), dim3(256), 0, stream>>>(
        (unsigned int*)ec, cw);
  }

  unsigned short* eh0 = eh_bf;
  unsigned short* eh1 = eh_bf + (size_t)BATCH * HE_DIM;
  float* ec0 = ec;
  float* ec1 = ec + (size_t)BATCH * HE_DIM;

  // ---- encoder: 256 sequential steps, 2 layers ----
  for (int t = 0; t < S_LEN; ++t) {
    const unsigned short* x = src_bf + (size_t)t * BATCH * FDIM;
    gemm(x,   FDIM,   wih0e, eh0, HE_DIM, whh0e, ebih0, ebhh0, zbuf, N4E);
    cell(zbuf, ec0, eh0, HE_DIM);
    gemm(eh0, HE_DIM, wih1e, eh1, HE_DIM, whh1e, ebih1, ebhh1, zbuf, N4E);
    cell(zbuf, ec1, eh1, HE_DIM);
  }

  // ---- ch = desc @ lin_W^T + lin_b ; widen hidden/cell ----
  gemm(desc_bf, DESC_DIM, linw_bf, nullptr, 0, nullptr, linb, nullptr, chbuf, CH_DIM);

  unsigned short* dh0 = dh_bf;
  unsigned short* dh1 = dh_bf + (size_t)BATCH * HD_DIM;
  float* dc0 = dc;
  float* dc1 = dc + (size_t)BATCH * HD_DIM;
  {
    int n = BATCH * HD_DIM;
    dim3 g((unsigned)((n + 255) / 256));
    k_build_dec_state<<<g, dim3(256), 0, stream>>>(chbuf, eh0, ec0, dh0, dc0);
    k_build_dec_state<<<g, dim3(256), 0, stream>>>(chbuf, eh1, ec1, dh1, dc1);
  }

  // ---- decoder: 128 autoregressive steps ----
  for (int t = 0; t < T_LEN; ++t) {
    const unsigned short* x =
        (t == 0) ? (src_bf + (size_t)(S_LEN - 1) * BATCH * FDIM) : xfb_bf;
    gemm(x,   FDIM,   wih0d, dh0, HD_DIM, whh0d, dbih0, dbhh0, zbuf, N4D);
    cell(zbuf, dc0, dh0, HD_DIM);
    gemm(dh0, HD_DIM, wih1d, dh1, HD_DIM, whh1d, dbih1, dbhh1, zbuf, N4D);
    cell(zbuf, dc1, dh1, HD_DIM);
    // out_t = h1 @ fc_W^T + fc_b  -> straight into d_out slice
    float* out_t = out + (size_t)t * BATCH * FDIM;
    gemm(dh1, HD_DIM, fcw_bf, nullptr, 0, nullptr, fcb, nullptr, out_t, FDIM);
    // feed back as next input (bf16)
    conv(out_t, xfb_bf, BATCH * FDIM);
  }
}
